// ChunkedResidualMambaBlock_39075612459682
// MI455X (gfx1250) — matrix operand order repaired and checked by reading, not compiled, not run on hardware
//
#include <hip/hip_runtime.h>
#include <hip/hip_bf16.h>

// ---------------- problem constants ----------------
#define BB   4
#define CC   96
#define HH   64
#define WW   64
#define BC   256          // chunks
#define LL   64           // tokens per chunk (8x8)
#define DNN  192          // 2*CC
#define KDIR 4
#define NST  16
#define DTR  6
#define XDC  38           // DTR + 2*NST
#define EPSF 1e-5f

typedef __attribute__((ext_vector_type(16))) __bf16 v16bf;
typedef __attribute__((ext_vector_type(8)))  float  v8f;
typedef int v4i_ __attribute__((vector_size(16)));

#define GLOBAL_AS __attribute__((address_space(1)))
#define LDS_AS    __attribute__((address_space(3)))

// workspace layout (float units, every region 32B-aligned)
#define OFF_STATS  ((size_t)0)                         // BB*CC*2
#define OFF_H      ((size_t)1024)                      // BC*LL*CC   (f32)
#define OFF_HLN    (OFF_H    + (size_t)BC*LL*CC)       // BC*LL*CC   (bf16, region oversized)
#define OFF_XZ     (OFF_HLN  + (size_t)BC*LL*CC)       // BC*LL*384  (f32)
#define OFF_XCONV  (OFF_XZ   + (size_t)BC*LL*384)      // BC*LL*DNN  (f32, token-major)
#define OFF_XDBL   (OFF_XCONV+ (size_t)BC*LL*DNN)      // BC*KDIR*LL*XDC (f32)
#define OFF_YS     (OFF_XDBL + (size_t)BC*KDIR*LL*XDC) // BC*KDIR*LL*DNN (f32)
#define OFF_YB     (OFF_YS   + (size_t)BC*KDIR*LL*DNN) // BC*LL*DNN  (bf16, oversized)
#define OFF_WIN    (OFF_YB   + (size_t)BC*LL*DNN)      // 24*3*32*16 bf16 fragments
#define OFF_WOUT   (OFF_WIN  + (size_t)18432)          // 6*6*32*16 bf16 fragments

#if defined(__has_builtin)
#if __has_builtin(__builtin_amdgcn_global_load_async_to_lds_b128) && \
    __has_builtin(__builtin_amdgcn_s_wait_asynccnt)
#define HAVE_ASYNC_LDS 1
#endif
#endif
#ifndef HAVE_ASYNC_LDS
#define HAVE_ASYNC_LDS 0
#endif

__device__ __forceinline__ float sigmoidf_(float x) { return 1.0f / (1.0f + __expf(-x)); }
__device__ __forceinline__ float siluf_(float x)    { return x * sigmoidf_(x); }
__device__ __forceinline__ float softplusf_(float x){ return (x > 20.0f) ? x : log1pf(__expf(x)); }

// direction map: k0: l ; k1: T(l) ; k2: 63-l ; k3: T(63-l)  (T = 8x8 transpose, involution)
__device__ __forceinline__ int dir_map(int k, int l) {
  int ll = (k & 2) ? (63 - l) : l;
  return (k & 1) ? (((ll & 7) << 3) | (ll >> 3)) : ll;
}

// stage n16*16 bytes from global to LDS (async DMA when available)
__device__ __forceinline__ void stage_lds_16B(void* lds_dst, const void* gsrc, int n16, int tid) {
#if HAVE_ASYNC_LDS
  for (int i = tid; i < n16; i += 256)
    __builtin_amdgcn_global_load_async_to_lds_b128(
        (GLOBAL_AS v4i_*)((char*)gsrc + (size_t)i * 16),
        (LDS_AS v4i_*)((char*)lds_dst + (size_t)i * 16), 0, 0);
  __builtin_amdgcn_s_wait_asynccnt(0);
#else
  const uint4* s = (const uint4*)gsrc;
  uint4* d = (uint4*)lds_dst;
  for (int i = tid; i < n16; i += 256) d[i] = s[i];
#endif
}

// ---------------- KP: pre-convert weights to bf16 WMMA-fragment layout ----------------
// layout: [nt][kb32][lane][e], B-frag element e of lane: K = kb32*32 + e + hi*16, N = nt*16 + ln
__global__ void kp_wfrag(const float* __restrict__ in_w, const float* __restrict__ out_w,
                         float* __restrict__ ws) {
  int i = blockIdx.x * 256 + threadIdx.x;
  __bf16* win  = (__bf16*)(ws + OFF_WIN);
  __bf16* wout = (__bf16*)(ws + OFF_WOUT);
  if (i < 24 * 3 * 32 * 16) {
    int e = i & 15, lane = (i >> 4) & 31, rest = i >> 9;
    int kb32 = rest % 3, nt = rest / 3;
    int hi = lane >> 4, ln = lane & 15;
    win[i] = (__bf16)in_w[(size_t)(kb32 * 32 + e + hi * 16) * 384 + nt * 16 + ln];
  }
  if (i < 6 * 6 * 32 * 16) {
    int e = i & 15, lane = (i >> 4) & 31, rest = i >> 9;
    int kb32 = rest % 6, nt = rest / 6;
    int hi = lane >> 4, ln = lane & 15;
    wout[i] = (__bf16)out_w[(size_t)(kb32 * 32 + e + hi * 16) * CC + nt * 16 + ln];
  }
}

// ---------------- K0: instance-norm stats over H*W per (b,c) ----------------
__global__ void k0_stats(const float* __restrict__ x, float* __restrict__ ws) {
  __shared__ float s0[256], s1[256];
  const int bc0 = blockIdx.x;
  const float* p = x + (size_t)bc0 * (HH * WW);
  float s = 0.f, q = 0.f;
  for (int i = threadIdx.x; i < HH * WW; i += 256) { float v = p[i]; s += v; q += v * v; }
  s0[threadIdx.x] = s; s1[threadIdx.x] = q; __syncthreads();
  for (int st = 128; st > 0; st >>= 1) {
    if (threadIdx.x < st) { s0[threadIdx.x] += s0[threadIdx.x + st]; s1[threadIdx.x] += s1[threadIdx.x + st]; }
    __syncthreads();
  }
  if (threadIdx.x == 0) {
    float mean = s0[0] / (float)(HH * WW);
    float var  = s1[0] / (float)(HH * WW) - mean * mean;
    ws[OFF_STATS + 2 * bc0]     = mean;
    ws[OFF_STATS + 2 * bc0 + 1] = rsqrtf(var + EPSF);
  }
}

// ---------------- K1: chunk + norm + SiLU -> buf_h ; LN1 -> buf_hln (bf16) ----------------
__global__ void k1_chunk_ln(const float* __restrict__ x,
                            const float* __restrict__ gamma, const float* __restrict__ beta,
                            const float* __restrict__ ln1_g, const float* __restrict__ ln1_b,
                            float* __restrict__ ws) {
  __shared__ float lact[LL * CC];
  __shared__ float tmean[LL], trstd[LL];
  const int bc = blockIdx.x;
  const int b = bc >> 6, ci = bc & 63, hni = ci >> 3, wni = ci & 7;
  float* bh = ws + OFF_H + (size_t)bc * LL * CC;
  __bf16* bhln = (__bf16*)(ws + OFF_HLN) + (size_t)bc * LL * CC;
  for (int idx = threadIdx.x; idx < LL * CC; idx += 256) {
    int l = idx / CC, c = idx % CC;
    int hin = l >> 3, win = l & 7;
    int Hi = hni * 8 + hin, Wi = wni * 8 + win;
    float v = x[(((size_t)b * CC + c) * HH + Hi) * WW + Wi];
    float mean = ws[OFF_STATS + 2 * (b * CC + c)];
    float rstd = ws[OFF_STATS + 2 * (b * CC + c) + 1];
    float n = gamma[c] * (v - mean) * rstd + beta[c];
    float a = siluf_(n);
    bh[idx] = a; lact[idx] = a;
  }
  __syncthreads();
  if (threadIdx.x < LL) {
    int l = threadIdx.x;
    float s = 0.f;
    for (int c = 0; c < CC; ++c) s += lact[l * CC + c];
    float m = s / (float)CC, q = 0.f;
    for (int c = 0; c < CC; ++c) { float d = lact[l * CC + c] - m; q += d * d; }
    tmean[l] = m; trstd[l] = rsqrtf(q / (float)CC + EPSF);
  }
  __syncthreads();
  for (int idx = threadIdx.x; idx < LL * CC; idx += 256) {
    int l = idx / CC, c = idx % CC;
    bhln[idx] = (__bf16)((lact[idx] - tmean[l]) * trstd[l] * ln1_g[c] + ln1_b[c]);
  }
}

// ---------------- K2: in_proj GEMM via bf16 WMMA: (64x96)@(96x384) per chunk ----------------
__global__ void k2_inproj(float* __restrict__ ws) {
  __shared__ __bf16 ldsA[LL * CC];   // 12KB
  const int bc = blockIdx.x;
  const __bf16* bhln = (const __bf16*)(ws + OFF_HLN) + (size_t)bc * LL * CC;
  const v16bf* wf = (const v16bf*)(ws + OFF_WIN);
  float* xz = ws + OFF_XZ + (size_t)bc * LL * 384;
  stage_lds_16B(ldsA, bhln, LL * CC / 8, threadIdx.x);
  __syncthreads();
  const int wave = threadIdx.x >> 5, lane = threadIdx.x & 31;
  const int hi = lane >> 4, ln = lane & 15;
  for (int t = wave; t < 4 * 24; t += 8) {
    int mt = t / 24, nt = t % 24;
    v8f acc = {};
#pragma unroll
    for (int kb32 = 0; kb32 < 3; ++kb32) {
      v16bf a;
#pragma unroll
      for (int e = 0; e < 16; ++e) {
        int Ka = e + hi * 16;
        a[e] = ldsA[(mt * 16 + ln) * CC + kb32 * 32 + Ka];
      }
      v16bf bfr = wf[(nt * 3 + kb32) * 32 + lane];
      acc = __builtin_amdgcn_wmma_f32_16x16x32_bf16(false, a, false, bfr, (short)0, acc, false, false);
    }
#pragma unroll
    for (int r = 0; r < 8; ++r) {
      int m = mt * 16 + r + hi * 8;
      xz[(size_t)m * 384 + nt * 16 + ln] = acc[r];
    }
  }
}

// ---------------- K3: depthwise 3x3 conv + bias + SiLU -> xconv[bc][l][d] ----------------
__global__ void k3_dwconv(const float* __restrict__ cw, const float* __restrict__ cb,
                          float* __restrict__ ws) {
  int gid = blockIdx.x * 256 + threadIdx.x;
  if (gid >= BC * LL * DNN) return;
  int bc = gid / (LL * DNN), rem = gid % (LL * DNN);
  int l = rem / DNN, d = rem % DNN;
  int h0 = l >> 3, w0 = l & 7;
  const float* xzc = ws + OFF_XZ + (size_t)bc * LL * 384;
  float s = 0.f;
#pragma unroll
  for (int dh = -1; dh <= 1; ++dh) {
#pragma unroll
    for (int dw = -1; dw <= 1; ++dw) {
      int hh = h0 + dh, wwp = w0 + dw;
      if (hh >= 0 && hh < 8 && wwp >= 0 && wwp < 8)
        s += xzc[(size_t)(hh * 8 + wwp) * 384 + d] * cw[d * 9 + (dh + 1) * 3 + (dw + 1)];
    }
  }
  s += cb[d];
  ws[OFF_XCONV + (size_t)bc * LL * DNN + (size_t)l * DNN + d] = siluf_(s);
}

// ---------------- K4: x_proj: xdbl[bc][k][l][0..37] ----------------
__global__ void k4_xproj(const float* __restrict__ xpw, float* __restrict__ ws) {
  __shared__ float ldsX[LL * DNN];  // 48KB, token-major
  const int bid = blockIdx.x;
  const int bc = bid >> 2, k = bid & 3;
  const float* xc = ws + OFF_XCONV + (size_t)bc * LL * DNN;
  for (int idx = threadIdx.x; idx < LL * DNN; idx += 256) ldsX[idx] = xc[idx];
  __syncthreads();
  float* xd = ws + OFF_XDBL + ((size_t)bc * KDIR + k) * LL * XDC;
  for (int idx = threadIdx.x; idx < LL * XDC; idx += 256) {
    int l = idx / XDC, c = idx % XDC;
    int m = dir_map(k, l);
    const float* wr = xpw + ((size_t)k * XDC + c) * DNN;
    const float* xr = &ldsX[m * DNN];
    float s = 0.f;
    for (int d = 0; d < DNN; ++d) s = fmaf(xr[d], wr[d], s);
    xd[(size_t)l * XDC + c] = s;
  }
}

// ---------------- K5: selective scan, one lane per (bc,k,d) ----------------
__global__ void k5_scan(const float* __restrict__ dtw_g, const float* __restrict__ dtb_g,
                        const float* __restrict__ alog, const float* __restrict__ Ds,
                        float* __restrict__ ws) {
  int gid = blockIdx.x * 256 + threadIdx.x;
  if (gid >= BC * KDIR * DNN) return;
  int bc = gid / (KDIR * DNN), rem = gid % (KDIR * DNN);
  int k = rem / DNN, d = rem % DNN;
  int kd = k * DNN + d;
  float A[NST], h[NST], dtw[DTR];
#pragma unroll
  for (int n = 0; n < NST; ++n) { A[n] = -__expf(alog[(size_t)kd * NST + n]); h[n] = 0.f; }
#pragma unroll
  for (int r = 0; r < DTR; ++r) dtw[r] = dtw_g[(size_t)kd * DTR + r];
  const float dtb = dtb_g[kd];
  const float Dv = Ds[kd];
  const float* xcv = ws + OFF_XCONV + (size_t)bc * LL * DNN;     // [l][d]
  const float* xd  = ws + OFF_XDBL + ((size_t)bc * KDIR + k) * LL * XDC;
  float* ys = ws + OFF_YS + ((size_t)bc * KDIR + k) * LL * DNN;
  for (int l = 0; l < LL; ++l) {
    int m = dir_map(k, l);
    float u = xcv[(size_t)m * DNN + d];
    const float* row = xd + (size_t)l * XDC;
    float dt = dtb;
#pragma unroll
    for (int r = 0; r < DTR; ++r) dt = fmaf(row[r], dtw[r], dt);
    float delta = softplusf_(dt);
    float du = delta * u;
    float y = 0.f;
#pragma unroll
    for (int n = 0; n < NST; ++n) {
      float dA = __expf(delta * A[n]);
      h[n] = fmaf(h[n], dA, du * row[DTR + n]);
      y = fmaf(h[n], row[DTR + NST + n], y);
    }
    y = fmaf(Dv, u, y);
    ys[(size_t)m * DNN + d] = y;
  }
}

// ---------------- K6: sum 4 dirs + LN(192) + SiLU(z) gate -> yb (bf16) ----------------
__global__ void k6_combine(const float* __restrict__ og, const float* __restrict__ ob,
                           float* __restrict__ ws) {
  __shared__ float ldsY[LL * DNN];  // 48KB
  __shared__ float tmean[LL], trstd[LL];
  const int bc = blockIdx.x;
  const float* ys = ws + OFF_YS + (size_t)bc * KDIR * LL * DNN;
  for (int idx = threadIdx.x; idx < LL * DNN; idx += 256) {
    ldsY[idx] = ys[idx] + ys[idx + LL * DNN] + ys[idx + 2 * LL * DNN] + ys[idx + 3 * LL * DNN];
  }
  __syncthreads();
  if (threadIdx.x < LL) {
    int l = threadIdx.x;
    float s = 0.f;
    for (int c = 0; c < DNN; ++c) s += ldsY[l * DNN + c];
    float m = s / (float)DNN, q = 0.f;
    for (int c = 0; c < DNN; ++c) { float dd = ldsY[l * DNN + c] - m; q += dd * dd; }
    tmean[l] = m; trstd[l] = rsqrtf(q / (float)DNN + EPSF);
  }
  __syncthreads();
  const float* xz = ws + OFF_XZ + (size_t)bc * LL * 384;
  __bf16* yb = (__bf16*)(ws + OFF_YB) + (size_t)bc * LL * DNN;
  for (int idx = threadIdx.x; idx < LL * DNN; idx += 256) {
    int l = idx / DNN, c = idx % DNN;
    float v = (ldsY[idx] - tmean[l]) * trstd[l] * og[c] + ob[c];
    float z = xz[(size_t)l * 384 + DNN + c];
    yb[idx] = (__bf16)(v * siluf_(z));
  }
}

// ---------------- K7: out_proj WMMA (64x192)@(192x96) + residual + unchunk ----------------
__global__ void k7_outproj(const float* __restrict__ x, const float* __restrict__ alpha_p,
                           float* __restrict__ out, float* __restrict__ ws) {
  __shared__ __bf16 ldsA[LL * DNN];  // 24KB
  const int bc = blockIdx.x;
  const int b = bc >> 6, ci = bc & 63, hni = ci >> 3, wni = ci & 7;
  const float alpha = alpha_p[0];
  const __bf16* yb = (const __bf16*)(ws + OFF_YB) + (size_t)bc * LL * DNN;
  const float* bh = ws + OFF_H + (size_t)bc * LL * CC;
  const v16bf* wf = (const v16bf*)(ws + OFF_WOUT);
  stage_lds_16B(ldsA, yb, LL * DNN / 8, threadIdx.x);
  __syncthreads();
  const int wave = threadIdx.x >> 5, lane = threadIdx.x & 31;
  const int hi = lane >> 4, ln = lane & 15;
  for (int t = wave; t < 4 * 6; t += 8) {
    int mt = t / 6, nt = t % 6;
    v8f acc = {};
#pragma unroll
    for (int kb32 = 0; kb32 < 6; ++kb32) {
      v16bf a;
#pragma unroll
      for (int e = 0; e < 16; ++e) {
        int Ka = e + hi * 16;
        a[e] = ldsA[(mt * 16 + ln) * DNN + kb32 * 32 + Ka];
      }
      v16bf bfr = wf[(nt * 6 + kb32) * 32 + lane];
      acc = __builtin_amdgcn_wmma_f32_16x16x32_bf16(false, a, false, bfr, (short)0, acc, false, false);
    }
#pragma unroll
    for (int r = 0; r < 8; ++r) {
      int l = mt * 16 + r + hi * 8;
      int c = nt * 16 + ln;
      float xv = bh[(size_t)l * CC + c] + acc[r];   // h + ss2d(h)
      int hin = l >> 3, win = l & 7;
      int Hi = hni * 8 + hin, Wi = wni * 8 + win;
      size_t gi = (((size_t)b * CC + c) * HH + Hi) * WW + Wi;
      out[gi] = x[gi] + alpha * xv;
    }
  }
}

extern "C" void kernel_launch(void* const* d_in, const int* in_sizes, int n_in,
                              void* d_out, int out_size, void* d_ws, size_t ws_size,
                              hipStream_t stream) {
  const float* x        = (const float*)d_in[0];
  const float* gamma    = (const float*)d_in[1];
  const float* beta     = (const float*)d_in[2];
  const float* alpha    = (const float*)d_in[3];
  const float* ln1_g    = (const float*)d_in[4];
  const float* ln1_b    = (const float*)d_in[5];
  const float* in_proj  = (const float*)d_in[6];
  const float* conv_w   = (const float*)d_in[7];
  const float* conv_b   = (const float*)d_in[8];
  const float* x_proj_w = (const float*)d_in[9];
  const float* dt_w     = (const float*)d_in[10];
  const float* dt_b     = (const float*)d_in[11];
  const float* A_logs   = (const float*)d_in[12];
  const float* Ds       = (const float*)d_in[13];
  const float* onorm_g  = (const float*)d_in[14];
  const float* onorm_b  = (const float*)d_in[15];
  const float* out_proj = (const float*)d_in[16];
  float* out = (float*)d_out;
  float* ws  = (float*)d_ws;

  kp_wfrag   <<<(24 * 3 * 32 * 16 + 255) / 256, 256, 0, stream>>>(in_proj, out_proj, ws);
  k0_stats   <<<BB * CC, 256, 0, stream>>>(x, ws);
  k1_chunk_ln<<<BC, 256, 0, stream>>>(x, gamma, beta, ln1_g, ln1_b, ws);
  k2_inproj  <<<BC, 256, 0, stream>>>(ws);
  k3_dwconv  <<<(BC * LL * DNN + 255) / 256, 256, 0, stream>>>(conv_w, conv_b, ws);
  k4_xproj   <<<BC * KDIR, 256, 0, stream>>>(x_proj_w, ws);
  k5_scan    <<<(BC * KDIR * DNN + 255) / 256, 256, 0, stream>>>(dt_w, dt_b, A_logs, Ds, ws);
  k6_combine <<<BC, 256, 0, stream>>>(onorm_g, onorm_b, ws);
  k7_outproj <<<BC, 256, 0, stream>>>(x, alpha, out, ws);
}
